// DrugPairEncoder_65266323030537
// MI455X (gfx1250) — matrix operand order repaired and checked by reading, not compiled, not run on hardware
//
#include <hip/hip_runtime.h>
#include <hip/hip_bf16.h>

// ---------- problem constants (match reference) ----------
#define NN      50000
#define EE      800000
#define RR      8
#define DD      128
#define NDRUGS  4000
#define PP      8192
#define KCAT    1152          // R*D + D  (relation weights ++ root)

typedef __attribute__((ext_vector_type(16))) __bf16 v16bf;
typedef __attribute__((ext_vector_type(8)))  float  v8f;

__device__ __forceinline__ unsigned short f2bf(float f) {
    __bf16 b = (__bf16)f;                       // hw v_cvt to bf16
    return __builtin_bit_cast(unsigned short, b);
}
__device__ __forceinline__ float bf2f(unsigned short h) {
    return (float)__builtin_bit_cast(__bf16, h);
}

// packed-B fragment layout: Bp[((kt*2+half)*N + col)*16 + t],  k = kt*32 + half*16 + t
// -> one lane's 16-value WMMA B fragment is 32 contiguous bytes.

// ---------- trivial elementwise kernels ----------
__global__ void k_zero_f32(float* p, int n) {
    int i = blockIdx.x * 256 + threadIdx.x;
    if (i < n) p[i] = 0.f;
}
__global__ void k_zero_i32(int* p, int n) {
    int i = blockIdx.x * 256 + threadIdx.x;
    if (i < n) p[i] = 0;
}
__global__ void k_cvt_bf16(const float* src, unsigned short* dst, int n) {
    int i = blockIdx.x * 256 + threadIdx.x;
    if (i < n) dst[i] = f2bf(src[i]);
}

// counts per (dst, relation)
__global__ void k_count_edges(const int* ei, const int* et, float* cnt) {
    int e = blockIdx.x * 256 + threadIdx.x;
    if (e < EE) {
        int dst = ei[EE + e];
        int r   = et[e];
        unsafeAtomicAdd(&cnt[dst * RR + r], 1.0f);
    }
}
__global__ void k_invert_cnt(float* cnt, int n) {
    int i = blockIdx.x * 256 + threadIdx.x;
    if (i < n) cnt[i] = 1.0f / fmaxf(cnt[i], 1.0f);
}

// pack [W0..W7 | root] (K=128 x N=1152) into WMMA-fragment order
__global__ void k_pack_wcat(const float* W, const float* root, unsigned short* Bp) {
    int idx = blockIdx.x * 256 + threadIdx.x;
    if (idx >= DD * KCAT) return;
    int t    = idx & 15;
    int q    = idx >> 4;
    int col  = q % KCAT;
    int kh   = q / KCAT;
    int k    = (kh >> 1) * 32 + (kh & 1) * 16 + t;
    float v;
    if (col < RR * DD) {
        int r = col >> 7, o = col & 127;
        v = W[(r * DD + k) * DD + o];
    } else {
        v = root[k * DD + (col - RR * DD)];
    }
    Bp[idx] = f2bf(v);
}

// pack generic row-major f32 [K x N] into WMMA-fragment order
__global__ void k_pack_b(const float* B, unsigned short* Bp, int K, int N) {
    int idx = blockIdx.x * 256 + threadIdx.x;
    if (idx >= K * N) return;
    int t   = idx & 15;
    int q   = idx >> 4;
    int col = q % N;
    int kh  = q / N;
    int k   = (kh >> 1) * 32 + (kh & 1) * 16 + t;
    Bp[idx] = f2bf(B[k * N + col]);
}

// pack bil_W[o][i][j] as B[k=i*128+j][col=o] (K=16384, N=128) in fragment order
__global__ void k_pack_bilW(const float* bw, unsigned short* Bp) {
    int idx = blockIdx.x * 256 + threadIdx.x;
    if (idx >= 16384 * DD) return;
    int t   = idx & 15;
    int q   = idx >> 4;
    int col = q & 127;                  // N = 128
    int kh  = q >> 7;
    int k   = (kh >> 1) * 32 + (kh & 1) * 16 + t;
    int i = k >> 7, j = k & 127;
    Bp[idx] = f2bf(bw[(col * DD + i) * DD + j]);
}

// h[n][o] = (f32)Y[n][1024+o] + b[o]   (root contribution + bias)
__global__ void k_init_h(const unsigned short* Y, const float* b, float* h) {
    int i = blockIdx.x * 256 + threadIdx.x;
    if (i >= NN * DD) return;
    int n = i >> 7, o = i & 127;
    h[i] = bf2f(Y[n * KCAT + RR * DD + o]) + b[o];
}

// per edge: h[dst][:] += Y[src][etype*128 + :] * inv_cnt[dst,etype]
__global__ void k_scatter(const int* ei, const int* et, const unsigned short* Y,
                          const float* inv_cnt, float* h) {
    int e = blockIdx.x;
    int d = threadIdx.x;
    int src = ei[e];
    int dst = ei[EE + e];
    int r   = et[e];
    float inv = inv_cnt[dst * RR + r];
    float val = bf2f(Y[src * KCAT + r * DD + d]) * inv;
    unsafeAtomicAdd(&h[dst * DD + d], val);
}

// h = relu(h) in place; Xc = bf16(h)
__global__ void k_relu_cvt(float* h, unsigned short* Xc, int n) {
    int i = blockIdx.x * 256 + threadIdx.x;
    if (i < n) {
        float v = fmaxf(h[i], 0.f);
        h[i]  = v;
        Xc[i] = f2bf(v);
    }
}

// deterministic ordered compaction of node_type==0 (single block, 32 waves)
__global__ void k_drug_compact(const int* node_type, int* drug_idx) {
    __shared__ int wave_sums[32];
    __shared__ int base;
    int lane = threadIdx.x & 31;
    int wave = threadIdx.x >> 5;
    if (threadIdx.x == 0) base = 0;
    __syncthreads();
    for (int start = 0; start < NN; start += 1024) {
        int n = start + (int)threadIdx.x;
        int pred = (n < NN) && (node_type[n] == 0);
        unsigned long long mask = __ballot(pred);
        unsigned m32 = (unsigned)mask;
        int prefix = __popc(m32 & ((1u << lane) - 1u));
        if (lane == 0) wave_sums[wave] = __popc(m32);
        __syncthreads();
        int wprefix = 0, total = 0;
        for (int i = 0; i < 32; i++) {
            int s = wave_sums[i];
            if (i < wave) wprefix += s;
            total += s;
        }
        int g = base + wprefix + prefix;
        if (pred && g < NDRUGS) drug_idx[g] = n;
        __syncthreads();
        if (threadIdx.x == 0) base += total;
        __syncthreads();
    }
}

// fa/fb[p][d] (f32) = h[drug_idx[edge_filter[0/1][p]]][d]
__global__ void k_pair_gather(const float* h, const int* drug_idx, const int* ef,
                              float* fa, float* fb) {
    int i = blockIdx.x * 256 + threadIdx.x;
    if (i >= PP * DD) return;
    int p = i >> 7, d = i & 127;
    int ia = drug_idx[ef[p]];
    int ib = drug_idx[ef[PP + p]];
    fa[i] = h[ia * DD + d];
    fb[i] = h[ib * DD + d];
}

// ---------- generic WMMA GEMM ----------
// C[M x N] = act(A[M x K](bf16 row-major) @ Bp[K x N](bf16 packed-fragment) + bias)
// grid = (N/128, M/16), block = 256 (8 waves; wave w owns N-tile colBase = bx*128 + w*16)
__global__ void k_gemm_wmma(const unsigned short* __restrict__ A,
                            const unsigned short* __restrict__ Bp,
                            const float* bias, unsigned short* Cb, float* Cf,
                            int K, int Nn, int relu) {
    int mBase   = blockIdx.y * 16;
    int wave    = threadIdx.x >> 5;
    int lane    = threadIdx.x & 31;
    int colBase = blockIdx.x * 128 + wave * 16;
    int m     = lane & 15;
    int half  = (lane < 16) ? 0 : 1;
    int kbase = half * 16;              // A fragment K-offset (contiguous: k = 16*half + s)
    int col   = colBase + (lane & 15);
    const unsigned short* Arow = A + (size_t)(mBase + m) * K;
    v8f acc = {};
    for (int k0 = 0; k0 < K; k0 += 32) {
        union { v16bf v; uint4 q[2]; } Af, Bf;
        Af.q[0] = *(const uint4*)(Arow + k0 + kbase);        // k0+kbase .. +7
        Af.q[1] = *(const uint4*)(Arow + k0 + kbase + 8);    // k0+kbase+8 .. +15
        const unsigned short* bp =
            Bp + ((size_t)((k0 >> 5) * 2 + half) * Nn + col) * 16;
        Bf.q[0] = *(const uint4*)(bp);
        Bf.q[1] = *(const uint4*)(bp + 8);
        acc = __builtin_amdgcn_wmma_f32_16x16x32_bf16(false, Af.v, false, Bf.v,
                                                      (short)0, acc, false, false);
    }
#pragma unroll
    for (int r = 0; r < 8; r++) {
        int row = mBase + r + half * 8;
        float val = acc[r] + (bias ? bias[col] : 0.f);
        if (relu) val = fmaxf(val, 0.f);
        size_t o = (size_t)row * Nn + col;
        if (Cf) Cf[o] = val;
        if (Cb) Cb[o] = f2bf(val);
    }
}

// ---------- bilinear: out[p,o] = relu( sum_k (fa[p,i]*fb[p,j]) * W[k,o] + b[o] ), k=i*128+j
// A operand generated on the fly from 16-row LDS tiles; B prepacked in fragment order.
// grid = P/16 blocks, 256 threads (8 waves; wave w owns output cols w*16..w*16+15)
__global__ void k_bilinear_wmma(const float* __restrict__ fa, const float* __restrict__ fb,
                                const unsigned short* __restrict__ Bp,
                                const float* bias, unsigned short* out) {
    __shared__ float sfa[16][DD + 1];
    __shared__ float sfb[16][DD + 1];
    int pBase = blockIdx.x * 16;
    for (int t = threadIdx.x; t < 16 * DD; t += 256) {
        int pr = t >> 7, d = t & 127;
        sfa[pr][d] = fa[(pBase + pr) * DD + d];
        sfb[pr][d] = fb[(pBase + pr) * DD + d];
    }
    __syncthreads();
    int wave  = threadIdx.x >> 5;
    int lane  = threadIdx.x & 31;
    int m     = lane & 15;
    int half  = (lane < 16) ? 0 : 1;
    int kbase = half * 16;
    int col   = wave * 16 + (lane & 15);
    v8f acc = {};
    for (int k0 = 0; k0 < 16384; k0 += 32) {
        v16bf av;
        union { v16bf v; uint4 q[2]; } Bf;
#pragma unroll
        for (int vv = 0; vv < 8; vv++) {
            int k = k0 + kbase + 2 * vv;
            av[2 * vv]     = (__bf16)(sfa[m][k >> 7]       * sfb[m][k & 127]);
            av[2 * vv + 1] = (__bf16)(sfa[m][(k + 1) >> 7] * sfb[m][(k + 1) & 127]);
        }
        const unsigned short* bp =
            Bp + ((size_t)((k0 >> 5) * 2 + half) * DD + col) * 16;
        Bf.q[0] = *(const uint4*)(bp);
        Bf.q[1] = *(const uint4*)(bp + 8);
        acc = __builtin_amdgcn_wmma_f32_16x16x32_bf16(false, av, false, Bf.v,
                                                      (short)0, acc, false, false);
    }
#pragma unroll
    for (int r = 0; r < 8; r++) {
        int row = pBase + r + half * 8;
        float val = fmaxf(acc[r] + bias[col], 0.f);
        out[(size_t)row * DD + col] = f2bf(val);
    }
}

// ---------- host orchestration ----------
static inline size_t alignup(size_t x) { return (x + 255) & ~(size_t)255; }

extern "C" void kernel_launch(void* const* d_in, const int* in_sizes, int n_in,
                              void* d_out, int out_size, void* d_ws, size_t ws_size,
                              hipStream_t stream) {
    (void)in_sizes; (void)n_in; (void)out_size; (void)ws_size;
    const float* x    = (const float*)d_in[0];
    const int* ei     = (const int*)d_in[1];
    const int* et     = (const int*)d_in[2];
    const int* ntype  = (const int*)d_in[3];
    const int* ef     = (const int*)d_in[4];
    const float* W[3]    = { (const float*)d_in[5], (const float*)d_in[8],  (const float*)d_in[11] };
    const float* root[3] = { (const float*)d_in[6], (const float*)d_in[9],  (const float*)d_in[12] };
    const float* bL[3]   = { (const float*)d_in[7], (const float*)d_in[10], (const float*)d_in[13] };
    const float* bilW = (const float*)d_in[14];
    const float* bilB = (const float*)d_in[15];
    const float* mW[3] = { (const float*)d_in[16], (const float*)d_in[18], (const float*)d_in[20] };
    const float* mB[3] = { (const float*)d_in[17], (const float*)d_in[19], (const float*)d_in[21] };
    float* out = (float*)d_out;

    // workspace carve-out
    char* ws = (char*)d_ws;
    size_t off = 0;
    auto take = [&](size_t bytes) { char* p = ws + off; off += alignup(bytes); return p; };
    float*          cnt  = (float*)take((size_t)NN * RR * 4);
    unsigned short* Wcat = (unsigned short*)take((size_t)DD * KCAT * 2);
    unsigned short* Xc   = (unsigned short*)take((size_t)NN * DD * 2);
    unsigned short* Y    = (unsigned short*)take((size_t)NN * KCAT * 2);
    float*          h    = (float*)take((size_t)NN * DD * 4);
    int*            didx = (int*)take((size_t)NDRUGS * 4);
    float*          fa   = (float*)take((size_t)PP * DD * 4);
    float*          fb   = (float*)take((size_t)PP * DD * 4);
    unsigned short* Wt   = (unsigned short*)take((size_t)16384 * DD * 2);
    unsigned short* t0   = (unsigned short*)take((size_t)PP * DD * 2);
    unsigned short* t1   = (unsigned short*)take((size_t)PP * DD * 2);
    unsigned short* mpk[3];
    for (int i = 0; i < 3; i++) mpk[i] = (unsigned short*)take((size_t)DD * DD * 2);

    auto g1 = [](int n) { return dim3((n + 255) / 256); };

    // edge counts -> inverse means (shared by all 3 layers)
    k_zero_f32<<<g1(NN * RR), 256, 0, stream>>>(cnt, NN * RR);
    k_count_edges<<<g1(EE), 256, 0, stream>>>(ei, et, cnt);
    k_invert_cnt<<<g1(NN * RR), 256, 0, stream>>>(cnt, NN * RR);

    // layer-1 input
    k_cvt_bf16<<<g1(NN * DD), 256, 0, stream>>>(x, Xc, NN * DD);

    // --- 3 RGCN layers: transform-first (Y = Xc @ [W0..W7 | root]), scatter-mean, relu ---
    for (int L = 0; L < 3; L++) {
        k_pack_wcat<<<g1(DD * KCAT), 256, 0, stream>>>(W[L], root[L], Wcat);
        k_gemm_wmma<<<dim3(KCAT / 128, NN / 16), 256, 0, stream>>>(
            Xc, Wcat, nullptr, Y, nullptr, DD, KCAT, 0);
        k_init_h<<<g1(NN * DD), 256, 0, stream>>>(Y, bL[L], h);
        k_scatter<<<dim3(EE), DD, 0, stream>>>(ei, et, Y, cnt, h);
        k_relu_cvt<<<g1(NN * DD), 256, 0, stream>>>(h, Xc, NN * DD);
    }

    // --- drug index compaction + pair gather ---
    k_zero_i32<<<g1(NDRUGS), 256, 0, stream>>>(didx, NDRUGS);
    k_drug_compact<<<1, 1024, 0, stream>>>(ntype, didx);
    k_pair_gather<<<g1(PP * DD), 256, 0, stream>>>(h, didx, ef, fa, fb);

    // --- bilinear (K=16384 outer-product GEMM) ---
    k_pack_bilW<<<g1(16384 * DD), 256, 0, stream>>>(bilW, Wt);
    k_bilinear_wmma<<<dim3(PP / 16), 256, 0, stream>>>(fa, fb, Wt, bilB, t0);

    // --- MLP head ---
    for (int i = 0; i < 3; i++)
        k_pack_b<<<g1(DD * DD), 256, 0, stream>>>(mW[i], mpk[i], DD, DD);
    k_gemm_wmma<<<dim3(1, PP / 16), 256, 0, stream>>>(t0, mpk[0], mB[0], t1, nullptr, DD, DD, 1);
    k_gemm_wmma<<<dim3(1, PP / 16), 256, 0, stream>>>(t1, mpk[1], mB[1], t0, nullptr, DD, DD, 1);
    k_gemm_wmma<<<dim3(1, PP / 16), 256, 0, stream>>>(t0, mpk[2], mB[2], nullptr, out, DD, DD, 0);
}
